// MultiHeadLatentAttention_14224931685013
// MI455X (gfx1250) — compile-verified, matrix operand order checked by reading, not run on hardware
//
#include <hip/hip_runtime.h>
#include <hip/hip_bf16.h>
#include <math.h>

// ---------------------------------------------------------------------------
// MLA forward for MI455X (gfx1250, wave32, WMMA).
// All matmuls use v_wmma_f32_16x16x32_bf16 (bf16 operands, f32 accumulate).
// Pure-copy LDS staging uses GLOBAL_LOAD_ASYNC_TO_LDS_B128 (ASYNCcnt path).
// ---------------------------------------------------------------------------

typedef __bf16 bf16;
typedef __attribute__((ext_vector_type(16))) __bf16 v16bf;
typedef __attribute__((ext_vector_type(8)))  __bf16 v8bf;
typedef __attribute__((ext_vector_type(8)))  float  v8f;
typedef __attribute__((ext_vector_type(4)))  float  v4f;

__device__ __forceinline__ v16bf mk16(v8bf lo, v8bf hi) {
  v16bf r;
#pragma unroll
  for (int i = 0; i < 8; ++i) { r[i] = lo[i]; r[i + 8] = hi[i]; }
  return r;
}

__device__ __forceinline__ v8f wmma_bf16(v16bf a, v16bf b, v8f c) {
  // 8 args: (neg_a, A, neg_b, B, c_mod, C, reuse_a, reuse_b)
  return __builtin_amdgcn_wmma_f32_16x16x32_bf16(false, a, false, b,
                                                 (short)0, c, false, false);
}

// Flat pointer to __shared__ carries the wave-relative LDS byte offset in its
// low 32 bits (LDS aperture: addr[31:0] == LDS offset).
__device__ __forceinline__ unsigned lds_off(const void* p) {
  return (unsigned)(unsigned long long)p;
}

// Async copy 32 bytes global -> LDS (two b128 ops sharing one base address;
// INST_OFFSET applies to both the global and the LDS address).
__device__ __forceinline__ void async_copy32(unsigned laddr,
                                             const void* gaddr) {
  asm volatile(
      "global_load_async_to_lds_b128 %0, %1, off\n"
      "global_load_async_to_lds_b128 %0, %1, off offset:16"
      :
      : "v"(laddr), "v"((unsigned long long)gaddr)
      : "memory");
}

__device__ __forceinline__ void async_copy16(unsigned laddr,
                                             const void* gaddr) {
  asm volatile("global_load_async_to_lds_b128 %0, %1, off"
               :
               : "v"(laddr), "v"((unsigned long long)gaddr)
               : "memory");
}

__device__ __forceinline__ void wait_async0() {
#if __has_builtin(__builtin_amdgcn_s_wait_asynccnt)
  __builtin_amdgcn_s_wait_asynccnt(0);
#else
  asm volatile("s_wait_asynccnt 0x0" ::: "memory");
#endif
}

// ---------------------------------------------------------------------------
// Tiled GEMM: C[M,N] = A[M,K] * B[K,N]
//   A: bf16 (async-copied to LDS) or fp32 (converted while staging), row-major
//   B: fp32 weights, row-major, converted+transposed into LDS
//   OUT_MODE 0: fp32 row-major [M,N]
//   OUT_MODE 1: bf16 per-head layout [b, h, s, out_ld] at offset out_off
// Block tile 128x128, K-step 32, 256 threads = 8 waves.
// Wave w owns output rows [w*16, w*16+16) of the block tile (8 WMMA n-tiles).
// ---------------------------------------------------------------------------
template <typename TA, int OUT_MODE>
__global__ __launch_bounds__(256) void gemm_wmma(
    const TA* __restrict__ A, const float* __restrict__ B,
    void* __restrict__ outp, int M, int N, int K,
    int S, int H, int head_cols, int out_ld, int out_off) {
  __shared__ __align__(16) bf16 As[128][32];
  __shared__ __align__(16) bf16 Bt[128][32];  // B transposed: Bt[n][k]

  const int tid = threadIdx.x;
  const int w = tid >> 5, l = tid & 31;
  const int hh = l >> 4, mn = l & 15;
  const int rowBase = blockIdx.y * 128;
  const int colBase = blockIdx.x * 128;

  v8f acc[8];
#pragma unroll
  for (int j = 0; j < 8; ++j)
#pragma unroll
    for (int r = 0; r < 8; ++r) acc[j][r] = 0.0f;

  const int ar = tid >> 1;          // A stage: row 0..127
  const int ac = (tid & 1) * 16;    // A stage: col half
  const int bn = tid & 127;         // B stage: output col (Bt row)
  const int bk = (tid >> 7) * 16;   // B stage: k half
  const unsigned asLds = lds_off(&As[ar][ac]);

  for (int k0 = 0; k0 < K; k0 += 32) {
    // ---- stage A tile [128][32] ----
    {
      const TA* src = A + (size_t)(rowBase + ar) * K + k0 + ac;
      if constexpr (sizeof(TA) == 2) {
        // bf16 activations: pure copy -> async to LDS, no VGPR round-trip
        async_copy32(asLds, src);
      } else {
        v8f f0 = *(const v8f*)(src);
        v8f f1 = *(const v8f*)(src + 8);
        *(v8bf*)&As[ar][ac]     = __builtin_convertvector(f0, v8bf);
        *(v8bf*)&As[ar][ac + 8] = __builtin_convertvector(f1, v8bf);
      }
    }
    // ---- stage B tile transposed + fp32->bf16 convert ----
    {
      const float* src = B + (size_t)(k0 + bk) * N + colBase + bn;
      v8f f0, f1;
#pragma unroll
      for (int i = 0; i < 8; ++i) f0[i] = src[(size_t)i * N];
#pragma unroll
      for (int i = 0; i < 8; ++i) f1[i] = src[(size_t)(i + 8) * N];
      *(v8bf*)&Bt[bn][bk]     = __builtin_convertvector(f0, v8bf);
      *(v8bf*)&Bt[bn][bk + 8] = __builtin_convertvector(f1, v8bf);
      if (k0 + 32 < K) __builtin_prefetch(src + (size_t)32 * N, 0, 1);
    }
    if constexpr (sizeof(TA) == 2) wait_async0();
    __syncthreads();
    // ---- WMMA: one A-frag per wave, 8 B-frags ----
    v16bf a = mk16(*(const v8bf*)&As[w * 16 + mn][hh * 8],
                   *(const v8bf*)&As[w * 16 + mn][16 + hh * 8]);
#pragma unroll
    for (int j = 0; j < 8; ++j) {
      v16bf b = mk16(*(const v8bf*)&Bt[j * 16 + mn][hh * 8],
                     *(const v8bf*)&Bt[j * 16 + mn][16 + hh * 8]);
      acc[j] = wmma_bf16(a, b, acc[j]);
    }
    __syncthreads();
  }

  // ---- epilogue: C layout: lane holds col n=mn, rows M = hh*8 + r ----
#pragma unroll
  for (int j = 0; j < 8; ++j) {
    const int col = colBase + j * 16 + mn;
#pragma unroll
    for (int r = 0; r < 8; ++r) {
      const int row = rowBase + w * 16 + hh * 8 + r;
      float v = acc[j][r];
      if constexpr (OUT_MODE == 0) {
        ((float*)outp)[(size_t)row * N + col] = v;
      } else {
        const int head = col / head_cols;
        const int d    = col % head_cols;
        const int b_ = row / S, s_ = row % S;
        ((bf16*)outp)[(((size_t)(b_ * H + head)) * S + s_) * out_ld +
                      out_off + d] = (bf16)v;
      }
    }
  }
}

// ---------------------------------------------------------------------------
// Row LayerNorm: fp32 in -> bf16 out. One block (256 thr) per row.
// ---------------------------------------------------------------------------
__global__ __launch_bounds__(256) void layernorm_kernel(
    const float* __restrict__ in, const float* __restrict__ gamma,
    const float* __restrict__ beta, bf16* __restrict__ out, int width) {
  __shared__ float r1[256], r2[256];
  const int row = blockIdx.x, tid = threadIdx.x;
  const float* x = in + (size_t)row * width;
  float s = 0.f, s2 = 0.f;
  for (int i = tid; i < width; i += 256) {
    float v = x[i]; s += v; s2 += v * v;
  }
  r1[tid] = s; r2[tid] = s2;
  __syncthreads();
  for (int st = 128; st > 0; st >>= 1) {
    if (tid < st) { r1[tid] += r1[tid + st]; r2[tid] += r2[tid + st]; }
    __syncthreads();
  }
  const float mu  = r1[0] / (float)width;
  const float var = r2[0] / (float)width - mu * mu;
  const float rs  = rsqrtf(var + 1e-5f);
  bf16* y = out + (size_t)row * width;
  for (int i = tid; i < width; i += 256)
    y[i] = (bf16)((x[i] - mu) * rs * gamma[i] + beta[i]);
}

// ---------------------------------------------------------------------------
// RoPE in-place on the rope slice [off, off+64) of a [BH, S, ld] bf16 buffer.
// One thread per (bh, s, j<32) rotation pair.
// ---------------------------------------------------------------------------
__global__ void rope_kernel(bf16* __restrict__ buf, int S, int BH, int ld,
                            int off) {
  const int idx = blockIdx.x * blockDim.x + threadIdx.x;  // over BH*S*32
  const int j  = idx & 31;
  const int s  = (idx >> 5) % S;
  const int bh = idx / (32 * S);
  if (bh >= BH) return;
  bf16* p = buf + ((size_t)bh * S + s) * ld + off;
  // inv_freq[j] = 10000^(-j/32) = exp(-j * ln(10000)/32)
  const float f = __expf(-(float)j * (9.2103403719761840f / 32.0f));
  const float ang = (float)s * f;
  const float c = __cosf(ang), sn = __sinf(ang);
  const float x0 = (float)p[j], x1 = (float)p[j + 32];
  p[j]      = (bf16)(x0 * c - x1 * sn);
  p[j + 32] = (bf16)(x1 * c + x0 * sn);
}

// ---------------------------------------------------------------------------
// Flash attention: per block (q-tile 128, head h, batch b), 8 waves.
// d_qk = 192, d_v = 128, KV streamed in tiles of 64 with online softmax.
// K tile staged via async-to-LDS; QK^T and P*V via WMMA; P goes through
// wave-private LDS to switch from C-layout to A-layout (s_wait_dscnt).
// Output written bf16 row-major [b*S+s][h*128+d] to feed the final GEMM.
// ---------------------------------------------------------------------------
__global__ __launch_bounds__(256) void mla_attention(
    const bf16* __restrict__ q, const bf16* __restrict__ k,
    const bf16* __restrict__ v, bf16* __restrict__ out, int S, int H, int B) {
  __shared__ __align__(16) bf16 Ks[64][192];   // K tile (== B^T layout)
  __shared__ __align__(16) bf16 Vt[128][64];   // V tile transposed
  __shared__ __align__(16) bf16 Pb[8][16][64]; // wave-private P staging

  const int tid = threadIdx.x;
  const int w = tid >> 5, l = tid & 31;
  const int hh = l >> 4, mn = l & 15;
  const int qt = blockIdx.x, h = blockIdx.y, b = blockIdx.z;
  const int bh = b * H + h;
  const size_t qkBase = (size_t)bh * S * 192;
  const size_t vBase  = (size_t)bh * S * 128;
  const int qrow0 = qt * 128;

  // Q fragments for this wave's 16 rows, held in registers (6 x K=32)
  v16bf qf[6];
  {
    const bf16* qp = q + qkBase + (size_t)(qrow0 + w * 16 + mn) * 192;
#pragma unroll
    for (int kd = 0; kd < 6; ++kd)
      qf[kd] = mk16(*(const v8bf*)(qp + kd * 32 + hh * 8),
                    *(const v8bf*)(qp + kd * 32 + 16 + hh * 8));
  }

  float m_run[8], l_run[8];
  v8f o[8];
#pragma unroll
  for (int r = 0; r < 8; ++r) { m_run[r] = -1e30f; l_run[r] = 0.f; }
#pragma unroll
  for (int j = 0; j < 8; ++j)
#pragma unroll
    for (int r = 0; r < 8; ++r) o[j][r] = 0.f;

  const float scale = 0.07216878364870323f;  // 1/sqrt(192)

  for (int kv0 = 0; kv0 < S; kv0 += 64) {
    __syncthreads();
    // stage K tile [64][192]: 1536 16-byte chunks, 6 async copies per thread
    {
      const bf16* kp = k + qkBase + (size_t)kv0 * 192;
#pragma unroll
      for (int e = 0; e < 6; ++e) {
        const int vi = e * 256 + tid;
        const int kr = vi / 24;
        const int kc = (vi % 24) * 8;
        async_copy16(lds_off(&Ks[kr][kc]), kp + (size_t)kr * 192 + kc);
      }
    }
    // stage V tile transposed: Vt[dv][kv] (transpose -> stays on VALU path)
    {
      const int dv = tid & 127, kh = (tid >> 7) * 32;
      const bf16* vp = v + vBase + (size_t)(kv0 + kh) * 128 + dv;
#pragma unroll
      for (int i = 0; i < 32; ++i) Vt[dv][kh + i] = vp[(size_t)i * 128];
    }
    wait_async0();
    __syncthreads();

    // ---- scores S = Q K^T for 4 n-tiles of 16 kv positions ----
    v8f sc[4];
#pragma unroll
    for (int jn = 0; jn < 4; ++jn)
#pragma unroll
      for (int r = 0; r < 8; ++r) sc[jn][r] = 0.f;
#pragma unroll
    for (int kd = 0; kd < 6; ++kd) {
#pragma unroll
      for (int jn = 0; jn < 4; ++jn) {
        v16bf bb = mk16(*(const v8bf*)&Ks[jn * 16 + mn][kd * 32 + hh * 8],
                        *(const v8bf*)&Ks[jn * 16 + mn][kd * 32 + 16 + hh * 8]);
        sc[jn] = wmma_bf16(qf[kd], bb, sc[jn]);
      }
    }

    // ---- online softmax update (row stats per M = hh*8 + r) ----
#pragma unroll
    for (int r = 0; r < 8; ++r) {
      float mx = sc[0][r];
#pragma unroll
      for (int jn = 1; jn < 4; ++jn) mx = fmaxf(mx, sc[jn][r]);
      mx *= scale;
      mx = fmaxf(mx, __shfl_xor(mx, 1, 32));
      mx = fmaxf(mx, __shfl_xor(mx, 2, 32));
      mx = fmaxf(mx, __shfl_xor(mx, 4, 32));
      mx = fmaxf(mx, __shfl_xor(mx, 8, 32));
      const float mnew = fmaxf(m_run[r], mx);
      const float a = __expf(m_run[r] - mnew);
      float psum = 0.f;
#pragma unroll
      for (int jn = 0; jn < 4; ++jn) {
        const float p = __expf(sc[jn][r] * scale - mnew);
        sc[jn][r] = p;
        psum += p;
      }
      psum += __shfl_xor(psum, 1, 32);
      psum += __shfl_xor(psum, 2, 32);
      psum += __shfl_xor(psum, 4, 32);
      psum += __shfl_xor(psum, 8, 32);
      l_run[r] = l_run[r] * a + psum;
      m_run[r] = mnew;
#pragma unroll
      for (int j = 0; j < 8; ++j) o[j][r] *= a;
    }

    // ---- C-layout -> A-layout for P via wave-private LDS ----
#pragma unroll
    for (int jn = 0; jn < 4; ++jn)
#pragma unroll
      for (int r = 0; r < 8; ++r)
        Pb[w][hh * 8 + r][jn * 16 + mn] = (bf16)sc[jn][r];
    asm volatile("s_wait_dscnt 0x0" ::: "memory");

    // ---- O += P * V ----
#pragma unroll
    for (int kd = 0; kd < 2; ++kd) {
      v16bf ap = mk16(*(const v8bf*)&Pb[w][mn][kd * 32 + hh * 8],
                      *(const v8bf*)&Pb[w][mn][kd * 32 + 16 + hh * 8]);
#pragma unroll
      for (int j = 0; j < 8; ++j) {
        v16bf bv = mk16(*(const v8bf*)&Vt[j * 16 + mn][kd * 32 + hh * 8],
                        *(const v8bf*)&Vt[j * 16 + mn][kd * 32 + 16 + hh * 8]);
        o[j] = wmma_bf16(ap, bv, o[j]);
      }
    }
  }

  // ---- epilogue: normalize and write [b*S+s][h*128 + d] bf16 ----
#pragma unroll
  for (int r = 0; r < 8; ++r) {
    const float inv = 1.0f / l_run[r];
    const int srow = qrow0 + w * 16 + hh * 8 + r;
    const size_t obase = ((size_t)b * S + srow) * 2048 + h * 128;
#pragma unroll
    for (int j = 0; j < 8; ++j)
      out[obase + j * 16 + mn] = (bf16)(o[j][r] * inv);
  }
}

// ---------------------------------------------------------------------------
extern "C" void kernel_launch(void* const* d_in, const int* in_sizes, int n_in,
                              void* d_out, int out_size, void* d_ws,
                              size_t ws_size, hipStream_t stream) {
  const float* x        = (const float*)d_in[0];
  const float* Wq_down  = (const float*)d_in[1];
  const float* q_gamma  = (const float*)d_in[2];
  const float* q_beta   = (const float*)d_in[3];
  const float* Wq_up    = (const float*)d_in[4];
  const float* Wq_rope  = (const float*)d_in[5];
  const float* Wkv_down = (const float*)d_in[6];
  const float* kv_gamma = (const float*)d_in[7];
  const float* kv_beta  = (const float*)d_in[8];
  const float* Wk_up    = (const float*)d_in[9];
  const float* Wv_up    = (const float*)d_in[10];
  const float* Wk_rope  = (const float*)d_in[11];
  const float* Wout     = (const float*)d_in[12];
  (void)in_sizes; (void)n_in; (void)out_size; (void)ws_size;

  constexpr int Bb = 2, S = 2048, H = 16, TOK = Bb * S;  // 4096 tokens

  char* ws = (char*)d_ws;
  size_t off = 0;
  auto alloc = [&](size_t bytes) -> void* {
    void* p = ws + off;
    off += (bytes + 255) & ~(size_t)255;
    return p;
  };
  float* tmp_q  = (float*)alloc((size_t)TOK * 1536 * 4);   // x@Wq_down
  float* tmp_kv = (float*)alloc((size_t)TOK * 512 * 4);    // x@Wkv_down
  bf16*  cq     = (bf16*)alloc((size_t)TOK * 1536 * 2);    // LN(q latent)
  bf16*  ckv    = (bf16*)alloc((size_t)TOK * 512 * 2);     // LN(kv latent)
  bf16*  qb     = (bf16*)alloc((size_t)Bb * H * S * 192 * 2);
  bf16*  kb     = (bf16*)alloc((size_t)Bb * H * S * 192 * 2);
  bf16*  vb     = (bf16*)alloc((size_t)Bb * H * S * 128 * 2);
  bf16*  ab     = (bf16*)alloc((size_t)TOK * 2048 * 2);    // attn out

  const dim3 blk(256);

  // latent projections (fp32 A, fp32 out -> LN)
  gemm_wmma<float, 0><<<dim3(1536 / 128, TOK / 128), blk, 0, stream>>>(
      x, Wq_down, tmp_q, TOK, 1536, 2048, S, H, 0, 0, 0);
  gemm_wmma<float, 0><<<dim3(512 / 128, TOK / 128), blk, 0, stream>>>(
      x, Wkv_down, tmp_kv, TOK, 512, 2048, S, H, 0, 0, 0);

  layernorm_kernel<<<TOK, blk, 0, stream>>>(tmp_q, q_gamma, q_beta, cq, 1536);
  layernorm_kernel<<<TOK, blk, 0, stream>>>(tmp_kv, kv_gamma, kv_beta, ckv, 512);

  // up-projections, written directly into per-head [b,h,s,ld] bf16 buffers
  gemm_wmma<bf16, 1><<<dim3(2048 / 128, TOK / 128), blk, 0, stream>>>(
      cq, Wq_up, qb, TOK, 2048, 1536, S, H, 128, 192, 0);
  gemm_wmma<bf16, 1><<<dim3(1024 / 128, TOK / 128), blk, 0, stream>>>(
      cq, Wq_rope, qb, TOK, 1024, 1536, S, H, 64, 192, 128);
  gemm_wmma<bf16, 1><<<dim3(2048 / 128, TOK / 128), blk, 0, stream>>>(
      ckv, Wk_up, kb, TOK, 2048, 512, S, H, 128, 192, 0);
  gemm_wmma<bf16, 1><<<dim3(1024 / 128, TOK / 128), blk, 0, stream>>>(
      ckv, Wk_rope, kb, TOK, 1024, 512, S, H, 64, 192, 128);
  gemm_wmma<bf16, 1><<<dim3(2048 / 128, TOK / 128), blk, 0, stream>>>(
      ckv, Wv_up, vb, TOK, 2048, 512, S, H, 128, 128, 0);

  // RoPE on the [128,192) slice of q and k
  const int ropeThreads = Bb * H * S * 32;
  rope_kernel<<<ropeThreads / 256, blk, 0, stream>>>(qb, S, Bb * H, 192, 128);
  rope_kernel<<<ropeThreads / 256, blk, 0, stream>>>(kb, S, Bb * H, 192, 128);

  // flash attention -> bf16 row-major [tok][2048]
  mla_attention<<<dim3(S / 128, H, Bb), blk, 0, stream>>>(qb, kb, vb, ab, S, H,
                                                          Bb);

  // output projection -> fp32 d_out
  gemm_wmma<bf16, 0><<<dim3(2048 / 128, TOK / 128), blk, 0, stream>>>(
      ab, Wout, (float*)d_out, TOK, 2048, 2048, S, H, 0, 0, 0);
}
